// Downsample_PASA_Debug_84550726189326
// MI455X (gfx1250) — matrix operand order repaired and checked by reading, not compiled, not run on hardware
//
#include <hip/hip_runtime.h>
#include <hip/hip_bf16.h>

// ---------------------------------------------------------------------------
// PASA downsample:  sigma = normalize(clamp(BN(conv3x3(pad(x)))));
//                   out   = sum_taps(patches * sigma), stride-2
//   x:      [16,64,112,112] f32
//   conv_w: [9,64,3,3]      f32
//   gamma/beta: [9]         f32
//   out:    [16,64,56,56]   f32
// ---------------------------------------------------------------------------

#define NB   16
#define CIN  64
#define H    112
#define W    112
#define KOUT 9
#define KTOT 576            // CIN * 9 taps
#define HO   56
#define WO   56
#define BFRAG (18 * 32 * 16)   // packed B fragment elements (bf16)

typedef __attribute__((ext_vector_type(16))) __bf16          v16bf;
typedef __attribute__((ext_vector_type(8)))  float           v8f;
typedef __attribute__((ext_vector_type(8)))  unsigned short  us8;

union FragAB { v16bf v; us8 h[2]; };
union FragC  { v8f v; float f[8]; };

__device__ __forceinline__ unsigned short f2bf(float f) {
    union { float f; unsigned u; } v; v.f = f;
    unsigned r = v.u + 0x7FFFu + ((v.u >> 16) & 1u);   // round-to-nearest-even
    return (unsigned short)(r >> 16);
}
__device__ __forceinline__ int refl(int i, int n) {    // reflect pad (torch 'reflect')
    if (i < 0)      i = -i;
    if (i > n - 1)  i = 2 * (n - 1) - i;
    return i;
}

// ---------------------------------------------------------------------------
// Kernel 0a: zero the BN accumulation scratch (harness does not re-poison)
// ---------------------------------------------------------------------------
__global__ void pasa_zero_stats(float* stats) {
    if (threadIdx.x < 2 * KOUT) stats[threadIdx.x] = 0.0f;
}

// ---------------------------------------------------------------------------
// Kernel 0b: pack conv weights once into WMMA-lane-ordered bf16 B fragments.
//   packB[(step*32 + flane)*16 + e],  K = 32*step + ((flane>=16)?8:0) + ((e<8)?e:e+8),
//   K = tap*64 + cin  ->  B[K][N] = conv_w[N][cin][tap]  (N<9 else 0).
//   18 KB total: stays L2-resident; conv blocks then do a coalesced copy.
// ---------------------------------------------------------------------------
__global__ void pasa_pack_b(const float* __restrict__ wg,
                            unsigned short* __restrict__ packB) {
    int i = blockIdx.x * 256 + threadIdx.x;
    if (i >= BFRAG) return;
    int s   = i >> 9;                   // /512
    int rem = i & 511;
    int fl  = rem >> 4;
    int e   = rem & 15;
    int K   = 32 * s + ((fl >= 16) ? 8 : 0) + ((e < 8) ? e : e + 8);
    int tap = K >> 6;                   // K = tap*64 + c
    int c   = K & 63;
    int N   = fl & 15;
    float v = (N < KOUT) ? wg[N * KTOT + c * 9 + tap] : 0.0f;
    packB[i] = f2bf(v);
}

// ---------------------------------------------------------------------------
// Kernel 1: conv3x3 (reflect pad) as WMMA GEMM  +  BN partial sums
//   grid : 16 batches * 7 row-tiles * 7 col-tiles  (tile = 16x16 positions)
//   block: 128 threads = 4 wave32; wave w owns image rows y0+4w .. y0+4w+3,
//          i.e. FOUR 16(M=cols) x 16(N=out-ch) WMMA tiles sharing each B frag.
//   K ordering: K = tap*64 + cin  (A and B agree), so each 32-wide K step is
//   two contiguous 8xbf16 chunks per lane in LDS -> ds_load_b128 pairs.
//   Per step/wave: 2 B loads + 8 A loads -> 4 x v_wmma_f32_16x16x32_bf16.
// ---------------------------------------------------------------------------
__global__ __launch_bounds__(128)
void pasa_conv_wmma(const float* __restrict__ xg,
                    const unsigned short* __restrict__ packB,
                    float* __restrict__ sigma,     // [16,9,112,112]
                    float* __restrict__ stats) {   // [sum(9), sumsq(9)]
    __shared__ __attribute__((aligned(16))) unsigned short xs[18 * 18 * CIN]; // 41.5 KB
    __shared__ __attribute__((aligned(16))) unsigned short bs[BFRAG];         // 18   KB
    __shared__ float sstats[2 * KOUT];

    const int tid  = threadIdx.x;
    const int lane = tid & 31;
    const int wv   = tid >> 5;              // 0..3  -> group of 4 rows

    const int tile = blockIdx.x;            // 16 * 49
    const int n    = tile / 49;
    const int t2   = tile % 49;
    const int y0   = (t2 / 7) * 16;         // 16 rows per tile
    const int x0   = (t2 % 7) * 16;         // 16 cols per tile

    // ---- coalesced copy of pre-packed B fragments (b128 in, b128 out) ------
    {
        const us8* src = (const us8*)packB;
        #pragma unroll
        for (int i = 0; i < BFRAG / 8 / 128; ++i) {      // 9 vector chunks/thread
            int j = i * 128 + tid;
            *(us8*)&bs[j * 8] = src[j];
        }
    }

    // ---- stage input slab: rows y0-1..y0+16, cols x0-1..x0+16, 64 ch -------
    // LDS layout: xs[(r*18 + lc)*64 + c]  (channel innermost for A frags)
    for (int i = tid; i < CIN * 18 * 18; i += 128) {
        int c   = i / (18 * 18);
        int rem = i - c * (18 * 18);
        int r   = rem / 18;
        int lc  = rem - r * 18;
        int iy  = refl(y0 - 1 + r, H);
        int ix  = refl(x0 - 1 + lc, W);
        float v = xg[((n * CIN + c) * H + iy) * W + ix];
        xs[(r * 18 + lc) * CIN + c] = f2bf(v);
    }
    if (tid < 2 * KOUT) sstats[tid] = 0.0f;
    __syncthreads();

    // ---- 18-step WMMA accumulation, 4 M-tiles per wave ---------------------
    const int M     = lane & 15;            // A row = position column in tile
    const int kbase = (lane >> 4) << 3;     // A lane K sub-offset (0 or 8)
    FragC acc0, acc1, acc2, acc3;
    acc0.v = (v8f){}; acc1.v = (v8f){}; acc2.v = (v8f){}; acc3.v = (v8f){};

    #pragma unroll
    for (int s = 0; s < 18; ++s) {
        const int tap = s >> 1;
        const int ty  = tap / 3, tx = tap - 3 * ty;
        const int c0  = ((s & 1) << 5) + kbase;           // channel base, chunk 0
        const int lc  = M + tx;                           // staged col
        const int rb  = 4 * wv + ty;                      // staged row, sub 0

        FragAB b;
        const int bx = (s * 32 + lane) * 16;
        b.h[0] = *(const us8*)&bs[bx];
        b.h[1] = *(const us8*)&bs[bx + 8];

        FragAB a0, a1, a2, a3;
        const int ax = (rb * 18 + lc) * CIN + c0;
        a0.h[0] = *(const us8*)&xs[ax];                   // K: c0..c0+7
        a0.h[1] = *(const us8*)&xs[ax + 16];              // K: c0+16..c0+23
        a1.h[0] = *(const us8*)&xs[ax + 18 * CIN];
        a1.h[1] = *(const us8*)&xs[ax + 18 * CIN + 16];
        a2.h[0] = *(const us8*)&xs[ax + 36 * CIN];
        a2.h[1] = *(const us8*)&xs[ax + 36 * CIN + 16];
        a3.h[0] = *(const us8*)&xs[ax + 54 * CIN];
        a3.h[1] = *(const us8*)&xs[ax + 54 * CIN + 16];

        acc0.v = __builtin_amdgcn_wmma_f32_16x16x32_bf16(
            false, a0.v, false, b.v, (short)0, acc0.v, false, false);
        acc1.v = __builtin_amdgcn_wmma_f32_16x16x32_bf16(
            false, a1.v, false, b.v, (short)0, acc1.v, false, false);
        acc2.v = __builtin_amdgcn_wmma_f32_16x16x32_bf16(
            false, a2.v, false, b.v, (short)0, acc2.v, false, false);
        acc3.v = __builtin_amdgcn_wmma_f32_16x16x32_bf16(
            false, a3.v, false, b.v, (short)0, acc3.v, false, false);
    }

    // ---- store sigma + BN partial sums ------------------------------------
    const int N    = lane & 15;             // output channel
    const int half = lane >> 4;
    if (N < KOUT) {
        float ls = 0.0f, lq = 0.0f;
        const FragC* accs[4] = { &acc0, &acc1, &acc2, &acc3 };
        #pragma unroll
        for (int sub = 0; sub < 4; ++sub) {
            const int y = y0 + 4 * wv + sub;
            float* srow = &sigma[((n * KOUT + N) * H + y) * W + x0];
            #pragma unroll
            for (int r = 0; r < 8; ++r) {
                float v = accs[sub]->f[r];
                srow[r + 8 * half] = v;     // position col = x0 + r + 8*half
                ls += v; lq += v * v;
            }
        }
        atomicAdd(&sstats[N],        ls);   // ds_add_f32
        atomicAdd(&sstats[KOUT + N], lq);
    }
    __syncthreads();
    if (tid < 2 * KOUT) atomicAdd(&stats[tid], sstats[tid]);
}

// ---------------------------------------------------------------------------
// Kernel 2: finalize BN -> scale = gamma*rsqrt(var+eps), shift = beta - mean*scale
// ---------------------------------------------------------------------------
__global__ void pasa_bn_finalize(const float* __restrict__ stats,
                                 const float* __restrict__ gamma,
                                 const float* __restrict__ beta,
                                 float* __restrict__ scale,
                                 float* __restrict__ shift) {
    int k = threadIdx.x;
    if (k >= KOUT) return;
    const float cnt  = (float)(NB * H * W);
    float mean = stats[k] / cnt;
    float var  = stats[KOUT + k] / cnt - mean * mean;
    float sc   = gamma[k] * rsqrtf(var + 1e-5f);
    scale[k] = sc;
    shift[k] = beta[k] - mean * sc;
}

// ---------------------------------------------------------------------------
// Kernel 3: tap-normalize sigma at strided positions, weighted 3x3 combine,
//           stride-2 output.  grid: 16*56 (n,y2);  block 256.
// ---------------------------------------------------------------------------
__global__ __launch_bounds__(256)
void pasa_combine(const float* __restrict__ xg,
                  const float* __restrict__ sigma,
                  const float* __restrict__ scale,
                  const float* __restrict__ shift,
                  float* __restrict__ out) {
    __shared__ float wl[KOUT][WO];
    const int n  = blockIdx.x / HO;
    const int y2 = blockIdx.x % HO;
    const int y  = 2 * y2;
    const int tid = threadIdx.x;

    if (tid < WO) {
        const int x = 2 * tid;
        float v[KOUT], ssum = 0.0f;
        #pragma unroll
        for (int k = 0; k < KOUT; ++k) {
            float s = sigma[((n * KOUT + k) * H + y) * W + x];
            s = s * scale[k] + shift[k];
            s = fmaxf(s, 1e-4f);
            v[k] = s; ssum += s;
        }
        const float inv = 1.0f / ssum;
        #pragma unroll
        for (int k = 0; k < KOUT; ++k) wl[k][tid] = v[k] * inv;
    }
    __syncthreads();

    for (int e = tid; e < CIN * WO; e += 256) {
        const int c  = e / WO;
        const int x2 = e - c * WO;
        const int x  = 2 * x2;
        const float* xc = &xg[(n * CIN + c) * H * W];
        float acc = 0.0f;
        #pragma unroll
        for (int t = 0; t < 9; ++t) {
            const int ty = t / 3, tx = t - 3 * ty;
            const int iy = refl(y - 1 + ty, H);
            const int ix = refl(x - 1 + tx, W);
            acc += wl[t][x2] * xc[iy * W + ix];
        }
        out[((n * CIN + c) * HO + y2) * WO + x2] = acc;
    }
}

// ---------------------------------------------------------------------------
extern "C" void kernel_launch(void* const* d_in, const int* in_sizes, int n_in,
                              void* d_out, int out_size, void* d_ws, size_t ws_size,
                              hipStream_t stream) {
    const float* x      = (const float*)d_in[0];
    const float* conv_w = (const float*)d_in[1];
    const float* gamma  = (const float*)d_in[2];
    const float* beta   = (const float*)d_in[3];
    float*       out    = (float*)d_out;

    // workspace layout (sigma, stats, packB sections 16-byte aligned)
    float* sigma = (float*)d_ws;                         // 16*9*112*112 = 1,806,336 f32
    float* stats = sigma + (size_t)NB * KOUT * H * W;    // sum[9] + sumsq[9]
    float* scale = stats + 2 * KOUT;                     // 9
    float* shift = scale + KOUT;                         // 9
    unsigned short* packB = (unsigned short*)(shift + KOUT); // 9216 bf16 (18 KB)

    pasa_zero_stats<<<1, 32, 0, stream>>>(stats);
    pasa_pack_b<<<(BFRAG + 255) / 256, 256, 0, stream>>>(conv_w, packB);
    pasa_conv_wmma<<<NB * 49, 128, 0, stream>>>(x, packB, sigma, stats);
    pasa_bn_finalize<<<1, 32, 0, stream>>>(stats, gamma, beta, scale, shift);
    pasa_combine<<<NB * HO, 256, 0, stream>>>(x, sigma, scale, shift, out);
}
